// OutfitGAT_47794396070703
// MI455X (gfx1250) — matrix operand order repaired and checked by reading, not compiled, not run on hardware
//
#include <hip/hip_runtime.h>
#include <hip/hip_bf16.h>
#include <stdint.h>

#define NN      50000
#define NE      800000
#define NET     (NE + NN)        /* edges + self loops = 850000 */
#define DIN     512
#define HID     256
#define NHEADS  4
#define HC      64
#define EMBD    128
#define NROOT   64

typedef __attribute__((ext_vector_type(16))) _Float16 v16h;
typedef __attribute__((ext_vector_type(8)))  _Float16 v8h;
typedef __attribute__((ext_vector_type(8)))  float    v8f;

// ===================== WMMA GEMM =====================
// Fragment loaders per CDNA5 ISA 7.12.2 (wave32, 16-bit, 16x16x32):
//  A (16x32): lane L holds row M=L&15; halves 0..7 = K[koff..koff+7],
//             halves 8..15 = K[koff+16..koff+23], koff = (L>=16)?8:0  -> gap 16
//  B (32x16): lane L holds col N=L&15; halves 0..7 = K[koff..koff+7],
//             halves 8..15 = K[koff+16..koff+23], koff = (L>=16)?8:0  -> gap 16
__device__ __forceinline__ v16h load_frag(const _Float16* p, int gap) {
  v8h lo = *(const v8h*)p;
  v8h hi = *(const v8h*)(p + gap);
  v16h r;
#pragma unroll
  for (int i = 0; i < 8; ++i) { r[i] = lo[i]; r[i + 8] = hi[i]; }
  return r;
}

// C[M,N] = A[M,K] @ B[K,N], A row-major f16, Bt = B^T ([N,K] row-major) f16.
// block = 256 threads (8 waves); wave w -> rows [m0,m0+32), cols [n0,n0+64):
// 2 A-fragments (rows m0/m0+16) shared across 4 B-fragments -> 8 WMMA / K-step.
__global__ __launch_bounds__(256)
void gemm_wmma_f16(const _Float16* __restrict__ A,
                   const _Float16* __restrict__ Bt,
                   float* __restrict__ C,
                   int M, int N, int K) {
  const int wave = threadIdx.x >> 5;
  const int lane = threadIdx.x & 31;
  const int hw16 = lane >> 4;      // 0 or 1 (half-wave)
  const int lid  = lane & 15;
  const int m0 = blockIdx.x * 256 + wave * 32;
  const int n0 = blockIdx.y * 64;

  int arow0 = m0 + lid;       if (arow0 >= M) arow0 = M - 1;   // clamp; store guarded
  int arow1 = m0 + 16 + lid;  if (arow1 >= M) arow1 = M - 1;
  const _Float16* aptr0 = A  + (size_t)arow0 * K + hw16 * 8;
  const _Float16* aptr1 = A  + (size_t)arow1 * K + hw16 * 8;
  const _Float16* bptr0 = Bt + (size_t)(n0 +  0 + lid) * K + hw16 * 8;
  const _Float16* bptr1 = Bt + (size_t)(n0 + 16 + lid) * K + hw16 * 8;
  const _Float16* bptr2 = Bt + (size_t)(n0 + 32 + lid) * K + hw16 * 8;
  const _Float16* bptr3 = Bt + (size_t)(n0 + 48 + lid) * K + hw16 * 8;

  v8f acc00 = {}, acc01 = {}, acc02 = {}, acc03 = {};
  v8f acc10 = {}, acc11 = {}, acc12 = {}, acc13 = {};
  for (int k0 = 0; k0 < K; k0 += 32) {
    v16h a0 = load_frag(aptr0 + k0, 16);
    v16h a1 = load_frag(aptr1 + k0, 16);
    v16h b0 = load_frag(bptr0 + k0, 16);
    v16h b1 = load_frag(bptr1 + k0, 16);
    v16h b2 = load_frag(bptr2 + k0, 16);
    v16h b3 = load_frag(bptr3 + k0, 16);
    acc00 = __builtin_amdgcn_wmma_f32_16x16x32_f16(false, a0, false, b0, (short)0, acc00, false, false);
    acc01 = __builtin_amdgcn_wmma_f32_16x16x32_f16(false, a0, false, b1, (short)0, acc01, false, false);
    acc02 = __builtin_amdgcn_wmma_f32_16x16x32_f16(false, a0, false, b2, (short)0, acc02, false, false);
    acc03 = __builtin_amdgcn_wmma_f32_16x16x32_f16(false, a0, false, b3, (short)0, acc03, false, false);
    acc10 = __builtin_amdgcn_wmma_f32_16x16x32_f16(false, a1, false, b0, (short)0, acc10, false, false);
    acc11 = __builtin_amdgcn_wmma_f32_16x16x32_f16(false, a1, false, b1, (short)0, acc11, false, false);
    acc12 = __builtin_amdgcn_wmma_f32_16x16x32_f16(false, a1, false, b2, (short)0, acc12, false, false);
    acc13 = __builtin_amdgcn_wmma_f32_16x16x32_f16(false, a1, false, b3, (short)0, acc13, false, false);
  }

  // C/D layout: VGPR r -> row r (lanes 0-15), row r+8 (lanes 16-31)
  const int mstore0 = m0 + hw16 * 8;
  const int mstore1 = m0 + 16 + hw16 * 8;
#pragma unroll
  for (int r = 0; r < 8; ++r) {
    int mm = mstore0 + r;
    if (mm < M) {
      float* cp = C + (size_t)mm * N + n0 + lid;
      cp[0]  = acc00[r];
      cp[16] = acc01[r];
      cp[32] = acc02[r];
      cp[48] = acc03[r];
    }
    int mm1 = mstore1 + r;
    if (mm1 < M) {
      float* cp = C + (size_t)mm1 * N + n0 + lid;
      cp[0]  = acc10[r];
      cp[16] = acc11[r];
      cp[32] = acc12[r];
      cp[48] = acc13[r];
    }
  }
}

// ===================== prep / elementwise =====================
__global__ void wt_prep_kernel(const float* __restrict__ W, _Float16* __restrict__ Wt,
                               int K, int N) {   // W[K,N] -> Wt[N,K] f16
  int idx = blockIdx.x * 256 + threadIdx.x;
  if (idx >= K * N) return;
  int k = idx / N, n = idx - k * N;
  Wt[(size_t)n * K + k] = (_Float16)W[idx];
}

__global__ void cvt_f16_kernel(const float* __restrict__ X, _Float16* __restrict__ Y, int n) {
  int idx = blockIdx.x * 256 + threadIdx.x;
  if (idx < n) Y[idx] = (_Float16)X[idx];
}

__global__ void zero_f32_kernel(float* __restrict__ p, int n) {
  int idx = blockIdx.x * 256 + threadIdx.x;
  if (idx < n) p[idx] = 0.f;
}

__global__ void select_kernel(const float* __restrict__ item, const float* __restrict__ outfit,
                              const int* __restrict__ nt, float* __restrict__ h0, int n) {
  int idx = blockIdx.x * 256 + threadIdx.x;
  if (idx >= n) return;
  h0[idx] = (nt[idx >> 8] == 1) ? item[idx] : outfit[idx];
}

// ===================== reductions / layernorm =====================
__device__ __forceinline__ float wave_sum(float v) {
#pragma unroll
  for (int o = 16; o; o >>= 1) v += __shfl_down(v, o, 32);
  return v;
}
__device__ __forceinline__ float block_sum(float v, float* sh) {
  v = wave_sum(v);
  int w = threadIdx.x >> 5, l = threadIdx.x & 31, nw = blockDim.x >> 5;
  if (l == 0) sh[w] = v;
  __syncthreads();
  if (w == 0) {
    float t = (l < nw) ? sh[l] : 0.f;
    t = wave_sum(t);
    if (l == 0) sh[0] = t;
  }
  __syncthreads();
  float r = sh[0];
  __syncthreads();
  return r;
}

// out = act(ln(in + bias, g, b)); one block (width threads) per row
template <int ACT>   // 0 = none, 1 = relu
__global__ void ln_act_kernel(const float* __restrict__ in, const float* __restrict__ bias,
                              const float* __restrict__ g, const float* __restrict__ b,
                              float* __restrict__ out) {
  __shared__ float sh[8];
  const int W = blockDim.x, c = threadIdx.x;
  const size_t base = (size_t)blockIdx.x * W;
  float v = in[base + c] + (bias ? bias[c] : 0.f);
  float mean = block_sum(v, sh) / W;
  float d = v - mean;
  float var = block_sum(d * d, sh) / W;
  float y = d * rsqrtf(var + 1e-5f) * g[c] + b[c];
  if (ACT == 1) y = fmaxf(y, 0.f);
  out[base + c] = y;
}

// ===================== GAT =====================
__device__ __forceinline__ unsigned fenc(float f) {
  unsigned u = __float_as_uint(f);
  return (u & 0x80000000u) ? ~u : (u | 0x80000000u);
}
__device__ __forceinline__ float fdec(unsigned u) {
  return __uint_as_float((u & 0x80000000u) ? (u & 0x7FFFFFFFu) : ~u);
}
__device__ __forceinline__ void edge_ends(const int* __restrict__ ei, int e, int& s, int& d) {
  if (e < NE) { s = ei[e]; d = ei[NE + e]; } else { s = d = e - NE; }
}

__global__ void sd_kernel(const float* __restrict__ xw, const float* __restrict__ as,
                          const float* __restrict__ ad, float* __restrict__ s,
                          float* __restrict__ d) {
  int idx = blockIdx.x * 256 + threadIdx.x;
  if (idx >= NN * NHEADS) return;
  int node = idx >> 2, h = idx & 3;
  const float* xp = xw + (size_t)node * HID + h * HC;
  float ss = 0.f, dd = 0.f;
#pragma unroll 4
  for (int c = 0; c < HC; ++c) { float v = xp[c]; ss += v * as[h * HC + c]; dd += v * ad[h * HC + c]; }
  s[idx] = ss; d[idx] = dd;
}

__global__ void gat_init_kernel(unsigned* __restrict__ mxe, float* __restrict__ z) {
  int idx = blockIdx.x * 256 + threadIdx.x;
  if (idx >= NN * NHEADS) return;
  mxe[idx] = fenc(-__builtin_inff());
  z[idx] = 0.f;
}

__global__ void edge_e_kernel(const int* __restrict__ ei, const float* __restrict__ s,
                              const float* __restrict__ d, float* __restrict__ ebuf,
                              unsigned* __restrict__ mxe) {
  int idx = blockIdx.x * 256 + threadIdx.x;
  if (idx >= NET * NHEADS) return;
  int e = idx >> 2, h = idx & 3, sn, dn;
  edge_ends(ei, e, sn, dn);
  float v = s[sn * NHEADS + h] + d[dn * NHEADS + h];
  v = (v > 0.f) ? v : 0.2f * v;          // leaky_relu(0.2)
  ebuf[idx] = v;
  atomicMax(&mxe[dn * NHEADS + h], fenc(v));
}

__global__ void edge_p_kernel(const int* __restrict__ ei, float* __restrict__ ebuf,
                              const unsigned* __restrict__ mxe, float* __restrict__ z) {
  int idx = blockIdx.x * 256 + threadIdx.x;
  if (idx >= NET * NHEADS) return;
  int e = idx >> 2, h = idx & 3, sn, dn;
  edge_ends(ei, e, sn, dn);
  float p = __expf(ebuf[idx] - fdec(mxe[dn * NHEADS + h]));
  ebuf[idx] = p;                          // in-place: e -> p
  atomicAdd(&z[dn * NHEADS + h], p);
}

__global__ void scatter_kernel(const int* __restrict__ ei, const float* __restrict__ pbuf,
                               const float* __restrict__ z, const float* __restrict__ xw,
                               float* __restrict__ out) {
  int idx = blockIdx.x * 256 + threadIdx.x;   // over NET*256
  int e = idx >> 8, ch = idx & 255, h = ch >> 6;
  if (e >= NET) return;
  int sn, dn;
  edge_ends(ei, e, sn, dn);
  float alpha = pbuf[e * NHEADS + h] / (z[dn * NHEADS + h] + 1e-16f);
  atomicAdd(&out[(size_t)dn * HID + ch], xw[(size_t)sn * HID + ch] * alpha);
}

__global__ void bias_add_kernel(float* __restrict__ o, const float* __restrict__ b, int n) {
  int idx = blockIdx.x * 256 + threadIdx.x;
  if (idx < n) o[idx] += b[idx & 255];
}

// ===================== root-row tail =====================
__global__ void gather_kernel(const int* __restrict__ root, const float* __restrict__ h0,
                              const float* __restrict__ one, const float* __restrict__ two,
                              float* __restrict__ hr, float* __restrict__ onr,
                              float* __restrict__ twr) {
  int idx = blockIdx.x * 256 + threadIdx.x;
  if (idx >= NROOT * HID) return;
  int r = idx >> 8, c = idx & 255;
  size_t src = (size_t)root[r] * HID + c;
  hr[idx] = h0[src]; onr[idx] = one[src]; twr[idx] = two[src];
}

__global__ void lw_kernel(const float* __restrict__ hr, const float* __restrict__ lww,
                          const float* __restrict__ lwb, float* __restrict__ hw) {
  int r = threadIdx.x;
  if (r >= NROOT) return;
  float a0 = lwb[0], a1 = lwb[1];
  const float* hp = hr + (size_t)r * HID;
  for (int j = 0; j < HID; ++j) { float h = hp[j]; a0 += h * lww[2 * j]; a1 += h * lww[2 * j + 1]; }
  float m = fmaxf(a0, a1);
  float e0 = __expf(a0 - m), e1 = __expf(a1 - m), s = e0 + e1;
  hw[2 * r] = e0 / s; hw[2 * r + 1] = e1 / s;
}

__global__ void mix_kernel(const float* __restrict__ hw, const float* __restrict__ onr,
                           const float* __restrict__ twr, float* __restrict__ o) {
  int idx = blockIdx.x * 256 + threadIdx.x;
  if (idx >= NROOT * HID) return;
  int r = idx >> 8;
  o[idx] = hw[2 * r] * onr[idx] + hw[2 * r + 1] * twr[idx];
}

// gates = sigmoid(ln(fgout + fg_b, g, b)); out = elu(outA * gates)
__global__ void gate_kernel(const float* __restrict__ fgout, const float* __restrict__ fb,
                            const float* __restrict__ fg, const float* __restrict__ fbb,
                            const float* __restrict__ outA, float* __restrict__ outB) {
  __shared__ float sh[8];
  int c = threadIdx.x;
  size_t base = (size_t)blockIdx.x * HID;
  float v = fgout[base + c] + fb[c];
  float mean = block_sum(v, sh) / HID;
  float d = v - mean;
  float var = block_sum(d * d, sh) / HID;
  float y = d * rsqrtf(var + 1e-5f) * fg[c] + fbb[c];
  float gate = 1.f / (1.f + __expf(-y));
  float ov = outA[base + c] * gate;
  outB[base + c] = (ov > 0.f) ? ov : (__expf(ov) - 1.f);   // elu
}

__global__ void combine_kernel(const float* __restrict__ outB, const float* __restrict__ spr,
                               const float* __restrict__ spb, const float* __restrict__ hop,
                               float* __restrict__ outC) {
  int idx = blockIdx.x * 256 + threadIdx.x;
  if (idx >= NROOT * HID) return;
  outC[idx] = hop[0] * outB[idx] + hop[1] * (spr[idx] + spb[idx & 255]);
}

// emb = ln(fpout + fp_b, g, b); emb /= max(||emb||, 1e-12); out = emb / TEMP
__global__ void final_kernel(const float* __restrict__ fpout, const float* __restrict__ fb,
                             const float* __restrict__ fg, const float* __restrict__ fbb,
                             float* __restrict__ out) {
  __shared__ float sh[8];
  int c = threadIdx.x;
  size_t base = (size_t)blockIdx.x * EMBD;
  float v = fpout[base + c] + fb[c];
  float mean = block_sum(v, sh) / EMBD;
  float d = v - mean;
  float var = block_sum(d * d, sh) / EMBD;
  float y = d * rsqrtf(var + 1e-5f) * fg[c] + fbb[c];
  float n2 = block_sum(y * y, sh);
  float scale = 1.f / fmaxf(sqrtf(n2), 1e-12f);
  out[base + c] = y * scale * 10.f;      // / TEMP (0.1)
}

// ===================== host side =====================
static inline dim3 g1(long long n) { return dim3((unsigned)((n + 255) / 256)); }

extern "C" void kernel_launch(void* const* d_in, const int* in_sizes, int n_in,
                              void* d_out, int out_size, void* d_ws, size_t ws_size,
                              hipStream_t stream) {
  const float* x      = (const float*)d_in[0];
  const float* ie_w1  = (const float*)d_in[1];
  const float* ie_b1  = (const float*)d_in[2];
  const float* ie_g1  = (const float*)d_in[3];
  const float* ie_bb1 = (const float*)d_in[4];
  const float* ie_w2  = (const float*)d_in[5];
  const float* ie_b2  = (const float*)d_in[6];
  const float* ie_g2  = (const float*)d_in[7];
  const float* ie_bb2 = (const float*)d_in[8];
  const float* oe_w1  = (const float*)d_in[9];
  const float* oe_b1  = (const float*)d_in[10];
  const float* oe_g1  = (const float*)d_in[11];
  const float* oe_bb1 = (const float*)d_in[12];
  const float* oe_w2  = (const float*)d_in[13];
  const float* oe_b2  = (const float*)d_in[14];
  const float* oe_g2  = (const float*)d_in[15];
  const float* oe_bb2 = (const float*)d_in[16];
  const float* c1_w   = (const float*)d_in[17];
  const float* c1_as  = (const float*)d_in[18];
  const float* c1_ad  = (const float*)d_in[19];
  const float* c1_b   = (const float*)d_in[20];
  const float* c2_w   = (const float*)d_in[21];
  const float* c2_as  = (const float*)d_in[22];
  const float* c2_ad  = (const float*)d_in[23];
  const float* c2_b   = (const float*)d_in[24];
  const float* ln_g   = (const float*)d_in[25];
  const float* ln_b   = (const float*)d_in[26];
  const float* fg_w   = (const float*)d_in[27];
  const float* fg_b   = (const float*)d_in[28];
  const float* fg_g   = (const float*)d_in[29];
  const float* fg_bb  = (const float*)d_in[30];
  const float* sp_w   = (const float*)d_in[31];
  const float* sp_b   = (const float*)d_in[32];
  const float* fp_w   = (const float*)d_in[33];
  const float* fp_b   = (const float*)d_in[34];
  const float* fp_g   = (const float*)d_in[35];
  const float* fp_bb  = (const float*)d_in[36];
  const float* lw_w   = (const float*)d_in[37];
  const float* lw_b   = (const float*)d_in[38];
  const float* hop    = (const float*)d_in[39];
  const int*   ei     = (const int*)d_in[40];
  const int*   ntype  = (const int*)d_in[41];
  const int*   root   = (const int*)d_in[42];
  float* out = (float*)d_out;

  // ---- workspace bump allocator ----
  char* wp = (char*)d_ws;
  auto alloc = [&](size_t bytes) -> void* {
    void* r = (void*)wp;
    wp += (bytes + 255) & ~(size_t)255;
    return r;
  };
  _Float16* xh     = (_Float16*)alloc((size_t)NN * DIN * 2);
  _Float16* wt_ie1 = (_Float16*)alloc((size_t)DIN * HID * 2);
  _Float16* wt_ie2 = (_Float16*)alloc((size_t)HID * HID * 2);
  _Float16* wt_oe1 = (_Float16*)alloc((size_t)DIN * HID * 2);
  _Float16* wt_oe2 = (_Float16*)alloc((size_t)HID * HID * 2);
  _Float16* wt_c1  = (_Float16*)alloc((size_t)HID * HID * 2);
  _Float16* wt_c2  = (_Float16*)alloc((size_t)HID * HID * 2);
  _Float16* wt_fg  = (_Float16*)alloc((size_t)HID * HID * 2);
  _Float16* wt_sp  = (_Float16*)alloc((size_t)HID * HID * 2);
  _Float16* wt_fp  = (_Float16*)alloc((size_t)HID * EMBD * 2);
  float* tmp   = (float*)alloc((size_t)NN * HID * 4);
  float* bufA  = (float*)alloc((size_t)NN * HID * 4);   // item, later 'two'
  float* bufB  = (float*)alloc((size_t)NN * HID * 4);   // outfit, later 'xw'
  float* h0    = (float*)alloc((size_t)NN * HID * 4);
  float* one   = (float*)alloc((size_t)NN * HID * 4);
  _Float16* act16 = (_Float16*)alloc((size_t)NN * HID * 2);
  float* sbuf  = (float*)alloc((size_t)NN * NHEADS * 4);
  float* dbuf  = (float*)alloc((size_t)NN * NHEADS * 4);
  unsigned* mxe = (unsigned*)alloc((size_t)NN * NHEADS * 4);
  float* zbuf  = (float*)alloc((size_t)NN * NHEADS * 4);
  float* ebuf  = (float*)alloc((size_t)NET * NHEADS * 4);   // e, then p in-place
  float* hr    = (float*)alloc((size_t)NROOT * HID * 4);
  float* onr   = (float*)alloc((size_t)NROOT * HID * 4);
  float* twr   = (float*)alloc((size_t)NROOT * HID * 4);
  float* rtmp  = (float*)alloc((size_t)NROOT * HID * 4);
  float* outA  = (float*)alloc((size_t)NROOT * HID * 4);
  float* outB  = (float*)alloc((size_t)NROOT * HID * 4);
  float* spr   = (float*)alloc((size_t)NROOT * HID * 4);
  float* outC  = (float*)alloc((size_t)NROOT * HID * 4);
  float* rtmpe = (float*)alloc((size_t)NROOT * EMBD * 4);
  _Float16* a16r = (_Float16*)alloc((size_t)NROOT * HID * 2);
  float* hw    = (float*)alloc((size_t)NROOT * 2 * 4);

  auto gemm = [&](const _Float16* A, const _Float16* Bt, float* C, int M, int N, int K) {
    dim3 grid((M + 255) / 256, N / 64);
    gemm_wmma_f16<<<grid, 256, 0, stream>>>(A, Bt, C, M, N, K);
  };
  auto prep = [&](const float* W, _Float16* Wt, int K, int N) {
    wt_prep_kernel<<<g1((long long)K * N), 256, 0, stream>>>(W, Wt, K, N);
  };
  auto cvt = [&](const float* X, _Float16* Y, long long n) {
    cvt_f16_kernel<<<g1(n), 256, 0, stream>>>(X, Y, (int)n);
  };

  // ---- weight prep (fp32 -> transposed f16) ----
  prep(ie_w1, wt_ie1, DIN, HID);  prep(ie_w2, wt_ie2, HID, HID);
  prep(oe_w1, wt_oe1, DIN, HID);  prep(oe_w2, wt_oe2, HID, HID);
  prep(c1_w,  wt_c1,  HID, HID);  prep(c2_w,  wt_c2,  HID, HID);
  prep(fg_w,  wt_fg,  HID, HID);  prep(sp_w,  wt_sp,  HID, HID);
  prep(fp_w,  wt_fp,  HID, EMBD);
  cvt(x, xh, (long long)NN * DIN);

  // ---- item encoder ----
  gemm(xh, wt_ie1, tmp, NN, HID, DIN);
  ln_act_kernel<1><<<NN, HID, 0, stream>>>(tmp, ie_b1, ie_g1, ie_bb1, bufA);
  cvt(bufA, act16, (long long)NN * HID);
  gemm(act16, wt_ie2, tmp, NN, HID, HID);
  ln_act_kernel<1><<<NN, HID, 0, stream>>>(tmp, ie_b2, ie_g2, ie_bb2, bufA);

  // ---- outfit encoder ----
  gemm(xh, wt_oe1, tmp, NN, HID, DIN);
  ln_act_kernel<1><<<NN, HID, 0, stream>>>(tmp, oe_b1, oe_g1, oe_bb1, bufB);
  cvt(bufB, act16, (long long)NN * HID);
  gemm(act16, wt_oe2, tmp, NN, HID, HID);
  ln_act_kernel<1><<<NN, HID, 0, stream>>>(tmp, oe_b2, oe_g2, oe_bb2, bufB);

  // ---- node-type select ----
  select_kernel<<<g1((long long)NN * HID), 256, 0, stream>>>(bufA, bufB, ntype, h0, NN * HID);

  auto run_gat = [&](const float* xw, const float* as, const float* ad,
                     const float* bias, float* o) {
    sd_kernel<<<g1((long long)NN * NHEADS), 256, 0, stream>>>(xw, as, ad, sbuf, dbuf);
    gat_init_kernel<<<g1((long long)NN * NHEADS), 256, 0, stream>>>(mxe, zbuf);
    edge_e_kernel<<<g1((long long)NET * NHEADS), 256, 0, stream>>>(ei, sbuf, dbuf, ebuf, mxe);
    edge_p_kernel<<<g1((long long)NET * NHEADS), 256, 0, stream>>>(ei, ebuf, mxe, zbuf);
    zero_f32_kernel<<<g1((long long)NN * HID), 256, 0, stream>>>(o, NN * HID);
    scatter_kernel<<<g1((long long)NET * 256), 256, 0, stream>>>(ei, ebuf, zbuf, xw, o);
    bias_add_kernel<<<g1((long long)NN * HID), 256, 0, stream>>>(o, bias, NN * HID);
  };

  // ---- GAT layer 1 (h0 -> one) ----
  cvt(h0, act16, (long long)NN * HID);
  gemm(act16, wt_c1, bufB, NN, HID, HID);          // xw1 = bufB
  run_gat(bufB, c1_as, c1_ad, c1_b, one);

  // ---- GAT layer 2 (one -> two = bufA) ----
  cvt(one, act16, (long long)NN * HID);
  gemm(act16, wt_c2, bufB, NN, HID, HID);          // xw2 = bufB
  run_gat(bufB, c2_as, c2_ad, c2_b, bufA);

  // ---- tail on 64 root rows only ----
  gather_kernel<<<g1((long long)NROOT * HID), 256, 0, stream>>>(root, h0, one, bufA, hr, onr, twr);
  lw_kernel<<<1, NROOT, 0, stream>>>(hr, lw_w, lw_b, hw);
  mix_kernel<<<g1((long long)NROOT * HID), 256, 0, stream>>>(hw, onr, twr, rtmp);
  ln_act_kernel<0><<<NROOT, HID, 0, stream>>>(rtmp, (const float*)nullptr, ln_g, ln_b, outA);

  cvt(outA, a16r, (long long)NROOT * HID);
  gemm(a16r, wt_fg, rtmp, NROOT, HID, HID);
  gate_kernel<<<NROOT, HID, 0, stream>>>(rtmp, fg_b, fg_g, fg_bb, outA, outB);

  cvt(onr, a16r, (long long)NROOT * HID);
  gemm(a16r, wt_sp, spr, NROOT, HID, HID);
  combine_kernel<<<g1((long long)NROOT * HID), 256, 0, stream>>>(outB, spr, sp_b, hop, outC);

  cvt(outC, a16r, (long long)NROOT * HID);
  gemm(a16r, wt_fp, rtmpe, NROOT, EMBD, HID);
  final_kernel<<<NROOT, EMBD, 0, stream>>>(rtmpe, fp_b, fp_g, fp_bb, out);

  (void)in_sizes; (void)n_in; (void)out_size; (void)ws_size;
}
